// ResBlock_2_77137612636472
// MI455X (gfx1250) — 8-bit matrix operand order repaired and checked by reading, compiled, not run on hardware
//
#include <hip/hip_runtime.h>
#include <hip/hip_bf16.h>

typedef __attribute__((ext_vector_type(8))) int v8i;
typedef __attribute__((ext_vector_type(4))) unsigned int u32x4;
typedef __attribute__((ext_vector_type(8))) int i32x8;
typedef __attribute__((ext_vector_type(4))) int i32x4;

#if defined(__gfx1250__) && __has_builtin(__builtin_amdgcn_tensor_load_to_lds) && \
    __has_builtin(__builtin_amdgcn_s_wait_tensorcnt)
#define USE_TDM 1
#else
#define USE_TDM 0
#endif

#define N_   32
#define CIN  16
#define COUT 32
#define H_   256
#define W_   256
#define HO   128
#define WO   128

__global__ void k_pack_w(const float* __restrict__ W1, int* __restrict__ packedA) {
    int idx = blockIdx.x * blockDim.x + threadIdx.x;
    if (idx >= 2 * 3 * 32 * 8) return;
    int v    = idx & 7;
    int lane = (idx >> 3) & 31;
    int tc   = idx >> 8;
    int c    = tc % 3;
    int t    = tc / 3;

    int co      = t * 16 + (lane & 15);
    int g       = lane >> 4;
    int khw     = c * 4 + (v >> 1);
    int ci_base = g * 8 + (v & 1) * 4;

    unsigned int dw = 0u;
    if (khw < 9) {
        int kh = khw / 3, kw = khw % 3;
        for (int j = 0; j < 4; ++j) {
            int ci  = ci_base + j;
            float w = W1[((co * CIN + ci) * 3 + kh) * 3 + kw];
            int s   = (w > 0.f) ? 1 : ((w < 0.f) ? -1 : 0);
            dw |= ((unsigned int)(s & 0xFF)) << (8 * j);
        }
    }
    packedA[idx] = (int)dw;
}

#define LDS_COLS 130
#define ACT_N    (4 * LDS_COLS * CIN)

__global__ __launch_bounds__(256) void k_conv_fused(const float* __restrict__ x,
                                                    const int* __restrict__ packedA,
                                                    const float* __restrict__ Wsc,
                                                    float* __restrict__ ybuf,
                                                    float* __restrict__ sbuf) {
    __shared__ __align__(16) unsigned char lds_act[ACT_N];
    __shared__ float lds_raw[CIN * 2 * 128];
    __shared__ float lds_avg[64 * CIN];
    __shared__ __align__(16) int lds_A[2 * 3 * 32 * 8];
    __shared__ float lds_wsc[COUT * CIN];

    int tid = threadIdx.x;
    int blk = blockIdx.x;
    int wb  = blk & 1;
    int ho  = (blk >> 1) & (HO - 1);
    int n   = blk >> 8;
    int wo_base = wb * 64;

#if USE_TDM
    if (tid < 32) {
        unsigned           lds_off = (unsigned)(size_t)(void*)&lds_A[0];
        unsigned long long ga      = (unsigned long long)(size_t)packedA;
        u32x4 g0 = { 1u,
                     lds_off,
                     (unsigned)(ga & 0xFFFFFFFFu),
                     (unsigned)((ga >> 32) & 0x01FFFFFFu) | (2u << 30) };
        i32x8 g1 = { 0x00020000,
                     (int)((1536u & 0xFFFFu) << 16),
                     0,
                     (int)(1536u << 16),
                     0, 0, 0, 0 };
        i32x4 gz = { 0, 0, 0, 0 };
#if __clang_major__ >= 23
        i32x8 gz8 = { 0, 0, 0, 0, 0, 0, 0, 0 };
        __builtin_amdgcn_tensor_load_to_lds(g0, g1, gz, gz, gz8, 0);
#else
        __builtin_amdgcn_tensor_load_to_lds(g0, g1, gz, gz, 0);
#endif
        __builtin_amdgcn_s_wait_tensorcnt(0);
    }
#else
    for (int i = tid; i < 2 * 3 * 32 * 8; i += 256) lds_A[i] = packedA[i];
#endif
    for (int i = tid; i < COUT * CIN; i += 256) lds_wsc[i] = Wsc[i];

    int wi0 = 2 * wo_base - 1;
    for (int i = tid; i < ACT_N; i += 256) {
        int col = i % LDS_COLS;
        int rc  = i / LDS_COLS;
        int row = rc & 3;
        int ci  = rc >> 2;
        signed char val = 0;
        if (row < 3) {
            int hi = 2 * ho - 1 + row;
            int wi = wi0 + col;
            if (hi >= 0 && hi < H_ && wi >= 0 && wi < W_) {
                float xv = x[((n * CIN + ci) * H_ + hi) * (long)W_ + wi];
                val = (xv > 0.f) ? 1 : ((xv < 0.f) ? -1 : 0);
                if (row >= 1 && col >= 1 && col <= 128)
                    lds_raw[((ci * 2 + (row - 1)) << 7) + (col - 1)] = xv;
            }
        }
        lds_act[(row * LDS_COLS + col) * CIN + ci] = (unsigned char)val;
    }
    __syncthreads();

    for (int i = tid; i < 64 * CIN; i += 256) {
        int ci  = i & (CIN - 1);
        int pos = i >> 4;
        int r0  = (ci * 2) << 7;
        float av = 0.25f * (lds_raw[r0 + 2 * pos]       + lds_raw[r0 + 2 * pos + 1] +
                            lds_raw[r0 + 128 + 2 * pos] + lds_raw[r0 + 128 + 2 * pos + 1]);
        lds_avg[pos * CIN + ci] = av;
    }
    __syncthreads();

    int lane = tid & 31;
    int wv   = tid >> 5;
    int t    = wv & 1;
    int sub  = wv >> 1;
    int l15  = lane & 15;
    int g    = lane >> 4;
    int pcol = sub * 16 + l15;

    v8i acc = {};
    #pragma unroll
    for (int c = 0; c < 3; ++c) {
        v8i a, b;
        #pragma unroll
        for (int r = 0; r < 8; ++r)
            a[r] = lds_A[((t * 3 + c) * 32 + lane) * 8 + r];
        #pragma unroll
        for (int r = 0; r < 8; ++r) {
            int khw     = c * 4 + (r >> 1);
            int ci_base = g * 8 + (r & 1) * 4;
            int kh  = khw / 3, kw = khw - kh * 3;
            int col = 2 * pcol + kw;
            b[r] = *(const int*)&lds_act[(kh * LDS_COLS + col) * CIN + ci_base];
        }
        acc = __builtin_amdgcn_wmma_i32_16x16x64_iu8(true, a, true, b, acc,
                                                     false, false);
    }

    int wo = wo_base + pcol;
    #pragma unroll
    for (int r = 0; r < 8; ++r) {
        int co = t * 16 + g * 8 + r;
        ybuf[((n * COUT + co) * HO + ho) * (long)WO + wo] = (float)acc[r];
    }

    int pos  = tid & 63;
    int cob  = (tid >> 6) * 8;
    int wo2  = wo_base + pos;
    #pragma unroll
    for (int k = 0; k < 8; ++k) {
        int co = cob + k;
        float accs = 0.f;
        #pragma unroll
        for (int ci = 0; ci < CIN; ++ci)
            accs = fmaf(lds_wsc[co * CIN + ci], lds_avg[pos * CIN + ci], accs);
        sbuf[((n * COUT + co) * HO + ho) * (long)WO + wo2] = accs;
    }
}

__global__ __launch_bounds__(256) void k_stats(const float* __restrict__ ybuf,
                                               const float* __restrict__ sbuf,
                                               float* __restrict__ stats) {
    __shared__ float ssum[256], ssq[256];
    int b = blockIdx.x;
    const float* src = (b < COUT) ? ybuf : sbuf;
    int ch = b & (COUT - 1);
    float sum = 0.f, sq = 0.f;
    const int per_ch4 = (HO * WO) / 4;
    for (int i = threadIdx.x; i < N_ * per_ch4; i += 256) {
        int nn  = i >> 12;
        int off = i & (per_ch4 - 1);
        const float4 v = *(const float4*)&src[(((long)nn * COUT + ch) << 14) + off * 4];
        sum += (v.x + v.y) + (v.z + v.w);
        sq  += (v.x * v.x + v.y * v.y) + (v.z * v.z + v.w * v.w);
    }
    ssum[threadIdx.x] = sum; ssq[threadIdx.x] = sq;
    __syncthreads();
    for (int s = 128; s > 0; s >>= 1) {
        if (threadIdx.x < (unsigned)s) {
            ssum[threadIdx.x] += ssum[threadIdx.x + s];
            ssq[threadIdx.x]  += ssq[threadIdx.x + s];
        }
        __syncthreads();
    }
    if (threadIdx.x == 0) {
        const float inv_cnt = 1.f / (float)(N_ * HO * WO);
        float m = ssum[0] * inv_cnt;
        stats[b * 2]     = m;
        stats[b * 2 + 1] = ssq[0] * inv_cnt - m * m;
    }
}

__global__ __launch_bounds__(256) void k_finalize(const float* __restrict__ ybuf,
                                                  const float* __restrict__ sbuf,
                                                  const float* __restrict__ stats,
                                                  const float* __restrict__ g1,
                                                  const float* __restrict__ b1,
                                                  const float* __restrict__ g2,
                                                  const float* __restrict__ b2,
                                                  float* __restrict__ out) {
    const int total4 = (N_ * COUT * HO * WO) / 4;
    int idx4 = blockIdx.x * 256 + threadIdx.x;
    if (idx4 >= total4) return;
    long base = (long)idx4 * 4;
    int co = (int)((base >> 14) & (COUT - 1));

    float sc1 = g1[co] * rsqrtf(stats[co * 2 + 1] + 1e-5f);
    float sh1 = b1[co] - stats[co * 2] * sc1;
    float sc2 = g2[co] * rsqrtf(stats[64 + co * 2 + 1] + 1e-5f);
    float sh2 = b2[co] - stats[64 + co * 2] * sc2;

    const float4 y = *(const float4*)&ybuf[base];
    const float4 s = *(const float4*)&sbuf[base];
    float4 o;
    o.x = fmaf(y.x, sc1, sh1) + fmaf(s.x, sc2, sh2);
    o.y = fmaf(y.y, sc1, sh1) + fmaf(s.y, sc2, sh2);
    o.z = fmaf(y.z, sc1, sh1) + fmaf(s.z, sc2, sh2);
    o.w = fmaf(y.w, sc1, sh1) + fmaf(s.w, sc2, sh2);
    *(float4*)&out[base] = o;
}

extern "C" void kernel_launch(void* const* d_in, const int* in_sizes, int n_in,
                              void* d_out, int out_size, void* d_ws, size_t ws_size,
                              hipStream_t stream) {
    const float* x      = (const float*)d_in[0];
    const float* W1     = (const float*)d_in[1];
    const float* gamma1 = (const float*)d_in[2];
    const float* beta1  = (const float*)d_in[3];
    const float* Wsc    = (const float*)d_in[4];
    const float* gamma2 = (const float*)d_in[5];
    const float* beta2  = (const float*)d_in[6];
    float* out = (float*)d_out;

    char* ws = (char*)d_ws;
    float* stats   = (float*)ws;
    int*   packedA = (int*)(ws + 512);
    float* ybuf    = (float*)(ws + 8192);
    float* sbuf    = ybuf + (long)N_ * COUT * HO * WO;

    const int total = N_ * COUT * HO * WO;

    k_pack_w    <<<6, 256, 0, stream>>>(W1, packedA);
    k_conv_fused<<<N_ * HO * 2, 256, 0, stream>>>(x, packedA, Wsc, ybuf, sbuf);
    k_stats     <<<64, 256, 0, stream>>>(ybuf, sbuf, stats);
    k_finalize  <<<(total / 4 + 255) / 256, 256, 0, stream>>>(ybuf, sbuf, stats,
                                                              gamma1, beta1,
                                                              gamma2, beta2, out);
}
